// EdgePredictorGNN_62569083568345
// MI455X (gfx1250) — matrix operand order repaired and checked by reading, not compiled, not run on hardware
//
#include <hip/hip_runtime.h>

typedef _Float16 f16;
typedef __attribute__((ext_vector_type(16))) _Float16 v16h;
typedef __attribute__((ext_vector_type(8)))  float    v8f;
typedef __attribute__((ext_vector_type(4)))  unsigned v4u;
typedef __attribute__((ext_vector_type(8)))  int      v8i;
typedef __attribute__((ext_vector_type(4)))  int      v4i;

#define CH 64
#define TS 72   // padded LDS tile row stride in halves (144B = 36 dwords -> conflict-free b128)

// ---------------------------------------------------------------------------
// TDM descriptor builders (CDNA5 ISA §8.3/8.4).  One 16-row x 64-col f16 tile,
// LDS destination padded to TS-half row stride via pad_enable
// (pad_interval code 4 = 32 dwords = one 128B row, pad_amount code 3 = 4 dwords).
// ---------------------------------------------------------------------------
__device__ __forceinline__ v4u tdm_group0(const void* gaddr, unsigned lds_off,
                                          bool gather) {
  unsigned long long ga = (unsigned long long)gaddr;
  v4u g0;
  g0[0] = 1u | (gather ? (1u << 31) : 0u);            // count=1; gather_mode; 16-bit idx
  g0[1] = lds_off;                                     // lds_addr (bytes)
  g0[2] = (unsigned)ga;                                // global_addr[31:0]
  g0[3] = (unsigned)((ga >> 32) & 0x01FFFFFFu) | (2u << 30);  // addr[56:32] | type=2
  return g0;
}

__device__ __forceinline__ v8i tdm_group1_rows(int tile_rows, unsigned tensor_rows) {
  v8i g1;
  g1[0] = (1 << 16)            // data_size = 1 (2 bytes)
        | (1 << 20)            // pad_enable
        | (4 << 22)            // pad_interval: 32 dwords
        | (3 << 25);           // pad_amount: 4 dwords
  g1[1] = (int)(64u << 16);    // tensor_dim0 = 64 (bits 48..63)
  g1[2] = (int)((tensor_rows & 0xFFFFu) << 16);        // tensor_dim1 lo
  g1[3] = (int)((tensor_rows >> 16) | (64u << 16));    // tensor_dim1 hi | tile_dim0=64
  g1[4] = tile_rows;           // tile_dim1 (#rows / #gather indices); tile_dim2=0
  g1[5] = 64;                  // tensor_dim0_stride = 64 elements
  g1[6] = 0;
  g1[7] = 0;
  return g1;
}

// ---------------------------------------------------------------------------
// WMMA fragment helpers (contiguous k = 16*hid + s in both A and B, wave32)
// ---------------------------------------------------------------------------
__device__ __forceinline__ v16h load_afrag(const f16* T, int kc, int m_lo,
                                           int hid) {
  v16h a;
  const f16* base = T + m_lo * TS + 32 * kc + 16 * hid;
#pragma unroll
  for (int v = 0; v < 8; ++v) {
    int rel = 2 * v;
    a[2 * v]     = base[rel];
    a[2 * v + 1] = base[rel + 1];
  }
  return a;
}

__device__ __forceinline__ v16h load_bfrag(const f16* __restrict__ W, int nt,
                                           int kc, int m_lo, int hid) {
  v16h b;
  const f16* base = W + (32 * kc + 16 * hid) * CH + nt * 16 + m_lo;
#pragma unroll
  for (int v = 0; v < 8; ++v) {
    b[2 * v]     = base[(2 * v) * CH];
    b[2 * v + 1] = base[(2 * v + 1) * CH];
  }
  return b;
}

__device__ __forceinline__ v8f wmma16(v16h a, v16h b, v8f c) {
  return __builtin_amdgcn_wmma_f32_16x16x32_f16(false, a, false, b, (short)0,
                                                c, false, false);
}

// 16x64 @ 64x64 GEMM from one padded LDS tile.
__device__ __forceinline__ void mlp_gemm2(const f16* in, const v16h (&bw)[8],
                                          v8f (&acc)[4], int m_lo, int hid) {
#pragma unroll
  for (int nt = 0; nt < 4; ++nt)
#pragma unroll
    for (int i = 0; i < 8; ++i) acc[nt][i] = 0.0f;
#pragma unroll
  for (int kc = 0; kc < 2; ++kc) {
    v16h a = load_afrag(in, kc, m_lo, hid);
#pragma unroll
    for (int nt = 0; nt < 4; ++nt)
      acc[nt] = wmma16(a, bw[nt * 2 + kc], acc[nt]);
  }
}

// 16x128 @ 128x64 GEMM, K split across two padded LDS tiles (concat halves).
__device__ __forceinline__ void mlp_gemm_pair(const f16* in0, const f16* in1,
                                              const v16h (&bw)[16],
                                              v8f (&acc)[4], int m_lo, int hid) {
#pragma unroll
  for (int nt = 0; nt < 4; ++nt)
#pragma unroll
    for (int i = 0; i < 8; ++i) acc[nt][i] = 0.0f;
#pragma unroll
  for (int kc = 0; kc < 4; ++kc) {
    const f16* in = (kc < 2) ? in0 : in1;
    v16h a = load_afrag(in, kc & 1, m_lo, hid);
#pragma unroll
    for (int nt = 0; nt < 4; ++nt)
      acc[nt] = wmma16(a, bw[nt * 4 + kc], acc[nt]);
  }
}

// bias + PReLU epilogue into a padded f16 LDS tile.
__device__ __forceinline__ void epilogue_prelu(v8f (&acc)[4],
                                               const float* __restrict__ bias,
                                               float slope, f16* out, int m_lo,
                                               int hid) {
#pragma unroll
  for (int nt = 0; nt < 4; ++nt) {
    int n = nt * 16 + m_lo;
    float bn = bias[n];
#pragma unroll
    for (int r = 0; r < 8; ++r) {
      float v = acc[nt][r] + bn;
      v = fmaxf(v, 0.0f) + slope * fminf(v, 0.0f);
      out[(r + 8 * hid) * TS + n] = (f16)v;
    }
  }
}

// ---------------------------------------------------------------------------
// Per-edge 4-layer MLP (K=128->64->64->64->64).
// FIRSTROW: which edge_index row supplies the FIRST 64 features.
// MODE 0: atomic segment-sum into outp by dst.  MODE 1: store at edge row.
// Staging is done by the Tensor Data Mover in gather mode (16 row indices).
// ---------------------------------------------------------------------------
template <int FIRSTROW, int MODE>
__global__ __launch_bounds__(256) void edge_mlp_kernel(
    const f16* __restrict__ feat, int NR, const int* __restrict__ eidx, int E,
    const f16* __restrict__ w0, const f16* __restrict__ w1,
    const f16* __restrict__ w2, const f16* __restrict__ w3,
    const float* __restrict__ b0, const float* __restrict__ b1,
    const float* __restrict__ b2, const float* __restrict__ b3,
    const float* __restrict__ a0p, const float* __restrict__ a1p,
    const float* __restrict__ a2p, float* __restrict__ outp) {
  constexpr int WAVES = 8;
  __shared__ f16 sX0[WAVES][16 * TS];
  __shared__ f16 sX1[WAVES][16 * TS];
  __shared__ f16 sH[WAVES][16 * TS];
  __shared__ int sD[WAVES][16];

  const int lane = threadIdx.x & 31;
  const int wv = threadIdx.x >> 5;
  const int m_lo = lane & 15;
  const int hid = lane >> 4;
  const float s0 = *a0p, s1 = *a1p, s2 = *a2p;
  const v8i z8 = {0, 0, 0, 0, 0, 0, 0, 0};

  // Weight B-fragments pinned in VGPRs for the whole kernel.
  v16h bw0[16], bw1[8], bw2[8], bw3[8];
#pragma unroll
  for (int nt = 0; nt < 4; ++nt) {
#pragma unroll
    for (int kc = 0; kc < 4; ++kc)
      bw0[nt * 4 + kc] = load_bfrag(w0, nt, kc, m_lo, hid);
#pragma unroll
    for (int kc = 0; kc < 2; ++kc) {
      bw1[nt * 2 + kc] = load_bfrag(w1, nt, kc, m_lo, hid);
      bw2[nt * 2 + kc] = load_bfrag(w2, nt, kc, m_lo, hid);
      bw3[nt * 2 + kc] = load_bfrag(w3, nt, kc, m_lo, hid);
    }
  }

  f16* X0 = sX0[wv];
  f16* X1 = sX1[wv];
  f16* H = sH[wv];
  int* D = sD[wv];
  const unsigned ldsX0 = (unsigned)(unsigned long long)(void*)X0;
  const unsigned ldsX1 = (unsigned)(unsigned long long)(void*)X1;

  const int nTiles = E >> 4;
  const int nGroups = (nTiles + WAVES - 1) / WAVES;

  for (int g = blockIdx.x; g < nGroups; g += gridDim.x) {
    const int tile = g * WAVES + wv;
    const bool act = tile < nTiles;
    const int e0 = tile << 4;

    __syncthreads();  // previous iteration's LDS consumers are done
    if (act) {
      if (MODE == 0 && lane < 16) D[lane] = eidx[E + e0 + lane];
      // Pack 16 row indices per concat half; issue two TDM gathers.
      const int base0 = FIRSTROW * E + e0;
      const int base1 = (1 - FIRSTROW) * E + e0;
      v4i q2, q3, r2, r3;
#pragma unroll
      for (int j = 0; j < 4; ++j) {
        q2[j] = (eidx[base0 + 2 * j] & 0xFFFF) |
                ((eidx[base0 + 2 * j + 1] & 0xFFFF) << 16);
        q3[j] = (eidx[base0 + 8 + 2 * j] & 0xFFFF) |
                ((eidx[base0 + 8 + 2 * j + 1] & 0xFFFF) << 16);
        r2[j] = (eidx[base1 + 2 * j] & 0xFFFF) |
                ((eidx[base1 + 2 * j + 1] & 0xFFFF) << 16);
        r3[j] = (eidx[base1 + 8 + 2 * j] & 0xFFFF) |
                ((eidx[base1 + 8 + 2 * j + 1] & 0xFFFF) << 16);
      }
      __builtin_amdgcn_tensor_load_to_lds(tdm_group0(feat, ldsX0, true),
                                          tdm_group1_rows(16, (unsigned)NR),
                                          q2, q3, z8, 0);
      __builtin_amdgcn_tensor_load_to_lds(tdm_group0(feat, ldsX1, true),
                                          tdm_group1_rows(16, (unsigned)NR),
                                          r2, r3, z8, 0);
      // Prefetch next group's edge indices while TDM runs.
      const int e0n = e0 + gridDim.x * WAVES * 16;
      if (e0n < E) {
        __builtin_prefetch(&eidx[base0 + (e0n - e0)], 0, 1);
        __builtin_prefetch(&eidx[base1 + (e0n - e0)], 0, 1);
      }
      __builtin_amdgcn_s_wait_tensorcnt(0);
    }
    __syncthreads();
    if (act) {  // L1: concat tiles -> H
      v8f acc[4];
      mlp_gemm_pair(X0, X1, bw0, acc, m_lo, hid);
      epilogue_prelu(acc, b0, s0, H, m_lo, hid);
    }
    __syncthreads();
    if (act) {  // L2: H -> X0
      v8f acc[4];
      mlp_gemm2(H, bw1, acc, m_lo, hid);
      epilogue_prelu(acc, b1, s1, X0, m_lo, hid);
    }
    __syncthreads();
    if (act) {  // L3: X0 -> H
      v8f acc[4];
      mlp_gemm2(X0, bw2, acc, m_lo, hid);
      epilogue_prelu(acc, b2, s2, H, m_lo, hid);
    }
    __syncthreads();
    if (act) {  // L4: H -> global
      v8f acc[4];
      mlp_gemm2(H, bw3, acc, m_lo, hid);
#pragma unroll
      for (int nt = 0; nt < 4; ++nt) {
        int n = nt * 16 + m_lo;
        float bn = b3[n];
#pragma unroll
        for (int r = 0; r < 8; ++r) {
          float v = acc[nt][r] + bn;
          int m2 = r + 8 * hid;
          if (MODE == 0)
            unsafeAtomicAdd(&outp[D[m2] * CH + n], v);  // segment_sum by dst
          else
            outp[(e0 + m2) * CH + n] = v;
        }
      }
    }
  }
}

// ---------------------------------------------------------------------------
// Per-node 4-layer MLP (64->...->64); f16 input table staged by normal-mode TDM.
// ---------------------------------------------------------------------------
__global__ __launch_bounds__(256) void node_mlp_kernel(
    const f16* __restrict__ aggh, const f16* __restrict__ w0,
    const f16* __restrict__ w1, const f16* __restrict__ w2,
    const f16* __restrict__ w3, const float* __restrict__ b0,
    const float* __restrict__ b1, const float* __restrict__ b2,
    const float* __restrict__ b3, const float* __restrict__ a0p,
    const float* __restrict__ a1p, const float* __restrict__ a2p,
    float* __restrict__ outH, int N) {
  constexpr int WAVES = 8;
  __shared__ f16 sX[WAVES][16 * TS];
  __shared__ f16 sH[WAVES][16 * TS];

  const int lane = threadIdx.x & 31;
  const int wv = threadIdx.x >> 5;
  const int m_lo = lane & 15;
  const int hid = lane >> 4;
  const float s0 = *a0p, s1 = *a1p, s2 = *a2p;

  v16h bw0[8], bw1[8], bw2[8], bw3[8];
#pragma unroll
  for (int nt = 0; nt < 4; ++nt) {
#pragma unroll
    for (int kc = 0; kc < 2; ++kc) {
      bw0[nt * 2 + kc] = load_bfrag(w0, nt, kc, m_lo, hid);
      bw1[nt * 2 + kc] = load_bfrag(w1, nt, kc, m_lo, hid);
      bw2[nt * 2 + kc] = load_bfrag(w2, nt, kc, m_lo, hid);
      bw3[nt * 2 + kc] = load_bfrag(w3, nt, kc, m_lo, hid);
    }
  }

  f16* X = sX[wv];
  f16* H = sH[wv];
  const unsigned ldsX = (unsigned)(unsigned long long)(void*)X;
  const v4i z4 = {0, 0, 0, 0};
  const v8i z8 = {0, 0, 0, 0, 0, 0, 0, 0};

  const int nTiles = N >> 4;
  const int nGroups = (nTiles + WAVES - 1) / WAVES;

  for (int g = blockIdx.x; g < nGroups; g += gridDim.x) {
    const int tile = g * WAVES + wv;
    const bool act = tile < nTiles;
    const int n0 = tile << 4;

    __syncthreads();
    if (act) {  // DMA one 16x64 f16 tile into padded LDS
      __builtin_amdgcn_tensor_load_to_lds(
          tdm_group0(aggh + (unsigned long long)n0 * CH, ldsX, false),
          tdm_group1_rows(16, (unsigned)N), z4, z4, z8, 0);
      __builtin_amdgcn_s_wait_tensorcnt(0);
    }
    __syncthreads();
    if (act) {
      v8f acc[4];
      mlp_gemm2(X, bw0, acc, m_lo, hid);
      epilogue_prelu(acc, b0, s0, H, m_lo, hid);
    }
    __syncthreads();
    if (act) {
      v8f acc[4];
      mlp_gemm2(H, bw1, acc, m_lo, hid);
      epilogue_prelu(acc, b1, s1, X, m_lo, hid);
    }
    __syncthreads();
    if (act) {
      v8f acc[4];
      mlp_gemm2(X, bw2, acc, m_lo, hid);
      epilogue_prelu(acc, b2, s2, H, m_lo, hid);
    }
    __syncthreads();
    if (act) {
      v8f acc[4];
      mlp_gemm2(H, bw3, acc, m_lo, hid);
#pragma unroll
      for (int nt = 0; nt < 4; ++nt) {
        int n = nt * 16 + m_lo;
        float bn = b3[n];
#pragma unroll
        for (int r = 0; r < 8; ++r)
          outH[(n0 + r + 8 * hid) * CH + n] = acc[nt][r] + bn;
      }
    }
  }
}

// ---------------------------------------------------------------------------
// Graph-norm reductions + normalize, format conversion, zero-fill
// ---------------------------------------------------------------------------
__global__ void cvt_f32_f16_kernel(const float* __restrict__ s,
                                   f16* __restrict__ d, int n) {
  int i = blockIdx.x * blockDim.x + threadIdx.x;
  if (i < n) d[i] = (f16)s[i];
}

__global__ void zero_f32_kernel(float* __restrict__ p, int n) {
  int i = blockIdx.x * blockDim.x + threadIdx.x;
  if (i < n) p[i] = 0.0f;
}

__global__ void colsum_kernel(const float* __restrict__ h, int N,
                              float* __restrict__ sums) {
  int t = blockIdx.x * blockDim.x + threadIdx.x;
  int c = t & 63;
  int r0 = t >> 6;
  int rs = (gridDim.x * blockDim.x) >> 6;
  float acc = 0.0f;
  for (int r = r0; r < N; r += rs) acc += h[r * CH + c];
  unsafeAtomicAdd(&sums[c], acc);
}

__global__ void colsumsq_kernel(const float* __restrict__ h, int N,
                                const float* __restrict__ sums,
                                const float* __restrict__ ms,
                                float* __restrict__ vars) {
  int t = blockIdx.x * blockDim.x + threadIdx.x;
  int c = t & 63;
  int r0 = t >> 6;
  int rs = (gridDim.x * blockDim.x) >> 6;
  float sub = ms[c] * (sums[c] / (float)N);
  float acc = 0.0f;
  for (int r = r0; r < N; r += rs) {
    float v = h[r * CH + c] - sub;
    acc += v * v;
  }
  unsafeAtomicAdd(&vars[c], acc);
}

__global__ void normalize_kernel(const float* __restrict__ h, int N,
                                 const float* __restrict__ sums,
                                 const float* __restrict__ vars,
                                 const float* __restrict__ ms,
                                 const float* __restrict__ w,
                                 const float* __restrict__ b,
                                 f16* __restrict__ outH) {
  int i = blockIdx.x * blockDim.x + threadIdx.x;
  if (i < N * CH) {
    int c = i & 63;
    float invN = 1.0f / (float)N;
    float t = h[i] - ms[c] * (sums[c] * invN);
    float scale = rsqrtf(vars[c] * invN + 1e-5f);
    outH[i] = (f16)(t * scale * w[c] + b[c]);
  }
}

// ---------------------------------------------------------------------------
// Host-side orchestration
// ---------------------------------------------------------------------------
struct MLPIn {
  const float* W[4];
  const float* B[4];
  const float* A[3];
};

extern "C" void kernel_launch(void* const* d_in, const int* in_sizes, int n_in,
                              void* d_out, int out_size, void* d_ws,
                              size_t ws_size, hipStream_t stream) {
  const float* x = (const float*)d_in[0];
  const int* eidx = (const int*)d_in[1];
  const int N = in_sizes[0] / CH;
  const int E = in_sizes[1] / 2;

  auto getmlp = [&](int base) {
    MLPIn m;
    const int wo[4] = {0, 3, 6, 9};
    const int bo[4] = {1, 4, 7, 10};
    const int ao[3] = {2, 5, 8};
    for (int i = 0; i < 4; ++i) {
      m.W[i] = (const float*)d_in[base + wo[i]];
      m.B[i] = (const float*)d_in[base + bo[i]];
    }
    for (int i = 0; i < 3; ++i) m.A[i] = (const float*)d_in[base + ao[i]];
    return m;
  };
  MLPIn mlp[5] = {getmlp(2), getmlp(13), getmlp(24), getmlp(35), getmlp(46)};
  const float* n1w = (const float*)d_in[57];
  const float* n1b = (const float*)d_in[58];
  const float* n1ms = (const float*)d_in[59];
  const float* n2w = (const float*)d_in[60];
  const float* n2b = (const float*)d_in[61];
  const float* n2ms = (const float*)d_in[62];

  // Workspace carve-up
  char* ws = (char*)d_ws;
  size_t off = 0;
  auto carve = [&](size_t bytes) -> void* {
    void* p = ws + off;
    off = (off + bytes + 255) & ~(size_t)255;
    return p;
  };
  const int kin[5] = {128, 64, 128, 64, 128};  // msg/out/msg/out/edge
  f16* wf[5][4];
  for (int m = 0; m < 5; ++m)
    for (int i = 0; i < 4; ++i) {
      int k = (i == 0) ? kin[m] : CH;
      wf[m][i] = (f16*)carve((size_t)k * CH * sizeof(f16));
    }
  f16* xf16 = (f16*)carve((size_t)N * CH * sizeof(f16));
  f16* hf16 = (f16*)carve((size_t)N * CH * sizeof(f16));
  f16* aggh = (f16*)carve((size_t)N * CH * sizeof(f16));
  float* agg1 = (float*)carve((size_t)N * CH * sizeof(float));
  float* agg2 = (float*)carve((size_t)N * CH * sizeof(float));
  float* hbuf = (float*)carve((size_t)N * CH * sizeof(float));
  float* stats = (float*)carve(256 * sizeof(float));
  if (off > ws_size) return;  // workspace too small: refuse to corrupt memory

  // --- Prep: weight/activation conversion, zeroing -------------------------
  for (int m = 0; m < 5; ++m)
    for (int i = 0; i < 4; ++i) {
      int n = ((i == 0) ? kin[m] : CH) * CH;
      cvt_f32_f16_kernel<<<(n + 255) / 256, 256, 0, stream>>>(mlp[m].W[i],
                                                              wf[m][i], n);
    }
  cvt_f32_f16_kernel<<<(N * CH + 255) / 256, 256, 0, stream>>>(x, xf16, N * CH);
  zero_f32_kernel<<<(N * CH + 255) / 256, 256, 0, stream>>>(agg1, N * CH);
  zero_f32_kernel<<<(N * CH + 255) / 256, 256, 0, stream>>>(agg2, N * CH);
  zero_f32_kernel<<<1, 256, 0, stream>>>(stats, 256);

  const int edgeGroups = (E / 16 + 7) / 8;
  const int nodeGroups = (N / 16 + 7) / 8;

  // --- Conv1: msg (cat[h[dst],h[src]]) -> segment_sum -> node MLP ----------
  edge_mlp_kernel<1, 0><<<edgeGroups, 256, 0, stream>>>(
      xf16, N, eidx, E, wf[0][0], wf[0][1], wf[0][2], wf[0][3], mlp[0].B[0],
      mlp[0].B[1], mlp[0].B[2], mlp[0].B[3], mlp[0].A[0], mlp[0].A[1],
      mlp[0].A[2], agg1);
  cvt_f32_f16_kernel<<<(N * CH + 255) / 256, 256, 0, stream>>>(agg1, aggh,
                                                               N * CH);
  node_mlp_kernel<<<nodeGroups, 256, 0, stream>>>(
      aggh, wf[1][0], wf[1][1], wf[1][2], wf[1][3], mlp[1].B[0], mlp[1].B[1],
      mlp[1].B[2], mlp[1].B[3], mlp[1].A[0], mlp[1].A[1], mlp[1].A[2], hbuf, N);
  colsum_kernel<<<64, 256, 0, stream>>>(hbuf, N, stats + 0);
  colsumsq_kernel<<<64, 256, 0, stream>>>(hbuf, N, stats + 0, n1ms, stats + 64);
  normalize_kernel<<<(N * CH + 255) / 256, 256, 0, stream>>>(
      hbuf, N, stats + 0, stats + 64, n1ms, n1w, n1b, hf16);

  // --- Conv2 ----------------------------------------------------------------
  edge_mlp_kernel<1, 0><<<edgeGroups, 256, 0, stream>>>(
      hf16, N, eidx, E, wf[2][0], wf[2][1], wf[2][2], wf[2][3], mlp[2].B[0],
      mlp[2].B[1], mlp[2].B[2], mlp[2].B[3], mlp[2].A[0], mlp[2].A[1],
      mlp[2].A[2], agg2);
  cvt_f32_f16_kernel<<<(N * CH + 255) / 256, 256, 0, stream>>>(agg2, aggh,
                                                               N * CH);
  node_mlp_kernel<<<nodeGroups, 256, 0, stream>>>(
      aggh, wf[3][0], wf[3][1], wf[3][2], wf[3][3], mlp[3].B[0], mlp[3].B[1],
      mlp[3].B[2], mlp[3].B[3], mlp[3].A[0], mlp[3].A[1], mlp[3].A[2], hbuf, N);
  colsum_kernel<<<64, 256, 0, stream>>>(hbuf, N, stats + 128);
  colsumsq_kernel<<<64, 256, 0, stream>>>(hbuf, N, stats + 128, n2ms,
                                          stats + 192);
  normalize_kernel<<<(N * CH + 255) / 256, 256, 0, stream>>>(
      hbuf, N, stats + 128, stats + 192, n2ms, n2w, n2b, hf16);

  // --- Edge head: cat[h[row], h[col]] -> d_out [E,64] f32 -------------------
  edge_mlp_kernel<0, 1><<<edgeGroups, 256, 0, stream>>>(
      hf16, N, eidx, E, wf[4][0], wf[4][1], wf[4][2], wf[4][3], mlp[4].B[0],
      mlp[4].B[1], mlp[4].B[2], mlp[4].B[3], mlp[4].A[0], mlp[4].A[1],
      mlp[4].A[2], (float*)d_out);
}
